// GCN_33363305955881
// MI455X (gfx1250) — matrix operand order repaired and checked by reading, not compiled, not run on hardware
//
#include <hip/hip_runtime.h>

#define F_IN 165
#define KPAD 192   // 165 padded up to a multiple of 32 (WMMA K-depth)
#define HID  128
#define NOUT 2

typedef __attribute__((ext_vector_type(16))) _Float16 v16h;
typedef __attribute__((ext_vector_type(8)))  float    v8f;
typedef __attribute__((ext_vector_type(2)))  _Float16 h2v;

// ---------------- degree / normalization ----------------
__global__ void k_deg_init(float* deg, int n) {
    int i = blockIdx.x * blockDim.x + threadIdx.x;
    if (i < n) deg[i] = 1.0f;                       // self-loop contributes 1
}

__global__ void k_deg_edges(const int* __restrict__ dst, int e, float* deg) {
    int i = blockIdx.x * blockDim.x + threadIdx.x;
    if (i < e) atomicAdd(&deg[dst[i]], 1.0f);
}

__global__ void k_deg_rsqrt(float* deg, int n) {
    int i = blockIdx.x * blockDim.x + threadIdx.x;
    if (i < n) deg[i] = rsqrtf(deg[i]);             // in-place: deg -> dinv
}

// ---------------- GEMM1: h1 = x @ W1 (f16 WMMA, f32 accum) ----------------
// Block: 256 threads = 8 waves. Block tile: 128 rows x 128 cols.
// Wave w computes M-tile w (16 rows) x 8 N-tiles of 16 cols.
// Fused epilogue also emits a1 = b1 + h1 * dinv^2 (bias + self-loop term),
// saving a full 102 MB re-read pass over h1.
__global__ __launch_bounds__(256)
void k_gemm1_wmma(const float* __restrict__ x, const float* __restrict__ W1,
                  const float* __restrict__ b1, const float* __restrict__ dinv,
                  float* __restrict__ h1, float* __restrict__ a1, int n) {
    __shared__ _Float16 ldsA[128 * KPAD];   // [row][k]        48 KB
    __shared__ _Float16 ldsB[HID * KPAD];   // [col][k] (W1^T) 48 KB

    const int tid   = threadIdx.x;
    const int mbase = blockIdx.x * 128;

    // Stage A tile: convert f32 -> f16, zero-pad K 165..191 and OOB rows.
    for (int idx = tid; idx < 128 * KPAD; idx += 256) {
        int r = idx / KPAD, c = idx % KPAD;
        int row = mbase + r;
        float v = (c < F_IN && row < n) ? x[(long)row * F_IN + c] : 0.0f;
        ldsA[idx] = (_Float16)v;
    }
    // Stage W1 transposed: ldsB[col*KPAD + k] = W1[k*HID + col]
    for (int idx = tid; idx < HID * KPAD; idx += 256) {
        int c = idx / KPAD, k = idx % KPAD;
        float v = (k < F_IN) ? W1[k * HID + c] : 0.0f;
        ldsB[idx] = (_Float16)v;
    }
    __syncthreads();

    const int wave = tid >> 5;   // M-tile index 0..7
    const int lane = tid & 31;
    const int lrow = lane & 15;
    const int lhi  = lane >> 4;  // half-wave select

    v8f acc[8];
#pragma unroll
    for (int t = 0; t < 8; ++t) acc[t] = (v8f){0,0,0,0,0,0,0,0};

    const _Float16* arow = &ldsA[(wave * 16 + lrow) * KPAD];

    for (int k0 = 0; k0 < KPAD; k0 += 32) {
        // A fragment, 16-bit A 16x32 layout (ISA 7.12.2):
        // VGPR j<4: K = 2j + lhi*8 ; VGPR j>=4: K = 16 + 2(j-4) + lhi*8
        v16h afrag;
#pragma unroll
        for (int j = 0; j < 8; ++j) {
            int koff = (j < 4) ? (2 * j + lhi * 8) : (16 + 2 * (j - 4) + lhi * 8);
            h2v p = *(const h2v*)&arow[k0 + koff];
            afrag[2 * j]     = p[0];
            afrag[2 * j + 1] = p[1];
        }
#pragma unroll
        for (int nt = 0; nt < 8; ++nt) {
            // B fragment 32x16 (N=lrow), same K order as A:
            // VGPR j<4: K = 2j + lhi*8 ; VGPR j>=4: K = 16 + 2(j-4) + lhi*8
            const _Float16* bcol = &ldsB[(nt * 16 + lrow) * KPAD];
            v16h bfrag;
#pragma unroll
            for (int j = 0; j < 8; ++j) {
                h2v p = *(const h2v*)&bcol[k0 + ((j < 4) ? (2 * j + lhi * 8) : (16 + 2 * (j - 4) + lhi * 8))];
                bfrag[2 * j]     = p[0];
                bfrag[2 * j + 1] = p[1];
            }
            acc[nt] = __builtin_amdgcn_wmma_f32_16x16x32_f16(
                false, afrag, false, bfrag, (short)0, acc[nt], false, false);
        }
    }

    // C/D layout: VGPR r -> row (r + lhi*8), col = N-tile*16 + lrow
    const int row0 = mbase + wave * 16 + lhi * 8;

    if (mbase + 128 <= n) {
        // Interior fast path: no guards.
        float d2[8];
#pragma unroll
        for (int r = 0; r < 8; ++r) {
            float d = dinv[row0 + r];
            d2[r] = d * d;
        }
#pragma unroll
        for (int nt = 0; nt < 8; ++nt) {
            int col  = nt * 16 + lrow;
            float bb = b1[col];
#pragma unroll
            for (int r = 0; r < 8; ++r) {
                long off = (long)(row0 + r) * HID + col;
                float v  = acc[nt][r];
                h1[off] = v;
                a1[off] = bb + v * d2[r];
            }
        }
    } else {
        // Boundary block (at most one per launch).
#pragma unroll
        for (int nt = 0; nt < 8; ++nt) {
            int col  = nt * 16 + lrow;
            float bb = b1[col];
#pragma unroll
            for (int r = 0; r < 8; ++r) {
                int row = row0 + r;
                if (row < n) {
                    float d  = dinv[row];
                    long off = (long)row * HID + col;
                    float v  = acc[nt][r];
                    h1[off] = v;
                    a1[off] = bb + v * d * d;
                }
            }
        }
    }
}

// ---------------- layer-1 aggregation ----------------
// One wave per edge; 32 lanes stripe the 128 features.
__global__ __launch_bounds__(256)
void k_a1_edges(const int* __restrict__ src, const int* __restrict__ dst,
                const float* __restrict__ dinv, const float* __restrict__ h1,
                float* __restrict__ a1, int e) {
    int gw   = (blockIdx.x * blockDim.x + threadIdx.x) >> 5;
    int lane = threadIdx.x & 31;
    if (gw >= e) return;
    int s = src[gw], d = dst[gw];
    float nrm = dinv[s] * dinv[d];
    const float* hs = &h1[(long)s * HID];
    float*       ad = &a1[(long)d * HID];
#pragma unroll
    for (int j = 0; j < 4; ++j) {
        int f = lane + 32 * j;
        atomicAdd(&ad[f], hs[f] * nrm);
    }
}

// ---------------- GEMM2: h2 = relu(a1) @ W2 (wave-per-node reduction) ----
// Fused epilogue also writes out = b2 + h2 * dinv^2 (bias + self-loop term).
__global__ __launch_bounds__(256)
void k_gemm2(const float* __restrict__ a1, const float* __restrict__ W2,
             const float* __restrict__ b2, const float* __restrict__ dinv,
             float* __restrict__ h2, float* __restrict__ out, int n) {
    __shared__ float w2s[HID * NOUT];
    for (int i = threadIdx.x; i < HID * NOUT; i += 256) w2s[i] = W2[i];
    __syncthreads();

    int gw   = (blockIdx.x * blockDim.x + threadIdx.x) >> 5;
    int lane = threadIdx.x & 31;
    if (gw >= n) return;

    const float* row = &a1[(long)gw * HID];
    float acc0 = 0.0f, acc1 = 0.0f;
#pragma unroll
    for (int j = 0; j < 4; ++j) {
        int f = lane + 32 * j;
        float v = fmaxf(row[f], 0.0f);          // relu (dropout eval = identity)
        acc0 += v * w2s[f * NOUT + 0];
        acc1 += v * w2s[f * NOUT + 1];
    }
#pragma unroll
    for (int m = 16; m >= 1; m >>= 1) {
        acc0 += __shfl_xor(acc0, m, 32);
        acc1 += __shfl_xor(acc1, m, 32);
    }
    if (lane == 0) {
        float d  = dinv[gw];
        float d2 = d * d;
        h2[(long)gw * NOUT + 0]  = acc0;
        h2[(long)gw * NOUT + 1]  = acc1;
        out[(long)gw * NOUT + 0] = b2[0] + acc0 * d2;
        out[(long)gw * NOUT + 1] = b2[1] + acc1 * d2;
    }
}

// ---------------- layer-2 aggregation ----------------
__global__ void k_out_edges(const int* __restrict__ src, const int* __restrict__ dst,
                            const float* __restrict__ dinv, const float* __restrict__ h2,
                            float* __restrict__ out, int e) {
    int i = blockIdx.x * blockDim.x + threadIdx.x;
    if (i < e) {
        int s = src[i], d = dst[i];
        float nrm = dinv[s] * dinv[d];
        atomicAdd(&out[2 * d + 0], h2[2 * s + 0] * nrm);
        atomicAdd(&out[2 * d + 1], h2[2 * s + 1] * nrm);
    }
}

// ---------------- launch ----------------
extern "C" void kernel_launch(void* const* d_in, const int* in_sizes, int n_in,
                              void* d_out, int out_size, void* d_ws, size_t ws_size,
                              hipStream_t stream) {
    const float* x  = (const float*)d_in[0];
    const int*   ei = (const int*)  d_in[1];
    const float* W1 = (const float*)d_in[2];
    const float* b1 = (const float*)d_in[3];
    const float* W2 = (const float*)d_in[4];
    const float* b2 = (const float*)d_in[5];

    const int n = in_sizes[0] / F_IN;
    const int e = in_sizes[1] / 2;
    const int* src = ei;
    const int* dst = ei + e;

    float* ws   = (float*)d_ws;
    float* dinv = ws;                               // n
    float* h1   = dinv + n;                         // n*HID
    float* a1   = h1 + (size_t)n * HID;             // n*HID
    float* h2   = a1 + (size_t)n * HID;             // n*NOUT
    float* out  = (float*)d_out;

    dim3 blk(256);
    // dinv must be ready before gemm1 (fused self-loop term).
    k_deg_init <<<(n + 255) / 256, blk, 0, stream>>>(dinv, n);
    k_deg_edges<<<(e + 255) / 256, blk, 0, stream>>>(dst, e, dinv);
    k_deg_rsqrt<<<(n + 255) / 256, blk, 0, stream>>>(dinv, n);

    k_gemm1_wmma<<<(n + 127) / 128, blk, 0, stream>>>(x, W1, b1, dinv, h1, a1, n);

    k_a1_edges<<<(e + 7) / 8, blk, 0, stream>>>(src, dst, dinv, h1, a1, e);

    k_gemm2<<<(n + 7) / 8, blk, 0, stream>>>(a1, W2, b2, dinv, h2, out, n);

    k_out_edges<<<(e + 255) / 256, blk, 0, stream>>>(src, dst, dinv, h2, out, e);
}
